// ProbAttention_20779051778616
// MI455X (gfx1250) — matrix operand order repaired and checked by reading, not compiled, not run on hardware
//
#include <hip/hip_runtime.h>
#include <hip/hip_bf16.h>

typedef __bf16 bf16_t;
typedef __attribute__((ext_vector_type(16))) __bf16 v16bf;
typedef __attribute__((ext_vector_type(8)))  float  v8f;

constexpr int B_  = 128;
constexpr int A_  = 64;
constexpr int N_  = 2000;
constexpr int DIN = 128;
constexpr int H_  = 8;
constexpr int CN  = 64;   // n-chunk for attn1 (31 full chunks + remainder 16)

#if defined(__gfx1250__) && __has_builtin(__builtin_amdgcn_global_load_async_to_lds_b128) && __has_builtin(__builtin_amdgcn_s_wait_asynccnt)
#define USE_ASYNC_LDS 1
typedef int v4i_t __attribute__((vector_size(16)));
typedef __attribute__((address_space(1))) v4i_t* gptr128_t;  // global (AS1) int4*
typedef __attribute__((address_space(3))) v4i_t* lptr128_t;  // LDS (AS3) int4*
#else
#define USE_ASYNC_LDS 0
#endif

// ---------------- WMMA helpers (CDNA5 16x16x32 bf16, f32 accum) ----------------

__device__ __forceinline__ v8f zero8() {
  v8f z;
#pragma unroll
  for (int i = 0; i < 8; ++i) z[i] = 0.0f;
  return z;
}

__device__ __forceinline__ v8f wmma_bf(v16bf a, v16bf b, v8f c) {
  return __builtin_amdgcn_wmma_f32_16x16x32_bf16(false, a, false, b, (short)0, c, false, false);
}

// A-matrix frag (16x32): lane l -> row l&15, K = {kb..kb+7} U {kb+16..kb+23}, kb=(l>>4)*8
__device__ __forceinline__ v16bf load_a(const bf16_t* base, int ld, int lane) {
  int row = lane & 15;
  int kb  = (lane >> 4) * 8;
  const bf16_t* p = base + (size_t)row * ld + kb;
  v16bf v;
#pragma unroll
  for (int j = 0; j < 8; ++j) { v[j] = p[j]; v[8 + j] = p[16 + j]; }
  return v;
}

__device__ __forceinline__ v16bf load_a_f32(const float* base, int ld, int lane) {
  int row = lane & 15;
  int kb  = (lane >> 4) * 8;
  const float* p = base + (size_t)row * ld + kb;
  v16bf v;
#pragma unroll
  for (int j = 0; j < 8; ++j) { v[j] = (bf16_t)p[j]; v[8 + j] = (bf16_t)p[16 + j]; }
  return v;
}

// B-matrix frag (32x16): lane l -> col l&15, K = {kb..kb+7} U {kb+16..kb+23}, kb=(l>>4)*8
// Source X stored row-major as X[n'][k] (i.e. X = B^T rows contiguous in k).
__device__ __forceinline__ v16bf load_b_row(const bf16_t* base, int ld, int lane) {
  int c  = lane & 15;
  int kb = (lane >> 4) * 8;
  const bf16_t* p = base + (size_t)c * ld + kb;
  v16bf v;
#pragma unroll
  for (int j = 0; j < 8; ++j) { v[j] = p[j]; v[8 + j] = p[16 + j]; }
  return v;
}

__device__ __forceinline__ v16bf load_b_row_f32(const float* base, int ld, int lane) {
  int c  = lane & 15;
  int kb = (lane >> 4) * 8;
  const float* p = base + (size_t)c * ld + kb;
  v16bf v;
#pragma unroll
  for (int j = 0; j < 8; ++j) { v[j] = (bf16_t)p[j]; v[8 + j] = (bf16_t)p[16 + j]; }
  return v;
}

// ---------------- Precompute: W_qk[h] = norm1 * w_q_h^T @ w_k1_h (stored transposed) ----

__global__ __launch_bounds__(256) void precompute_qk(const float* wq, const float* wk1, bf16_t* WqkT) {
  const int h = blockIdx.x;
  __shared__ float sq[64][128];
  __shared__ float sk[64][128];
  for (int i = threadIdx.x; i < 64 * 128; i += 256) {
    sq[i >> 7][i & 127] = wq[(size_t)(h * 64) * 128 + i];
    sk[i >> 7][i & 127] = wk1[(size_t)(h * 64) * 128 + i];
  }
  __syncthreads();
  const float norm1 = 0.125f; // 1/sqrt(64)
  for (int o = threadIdx.x; o < 128 * 128; o += 256) {
    int i = o >> 7, j = o & 127;
    float s = 0.f;
#pragma unroll 8
    for (int e = 0; e < 64; ++e) s += sq[e][i] * sk[e][j];
    WqkT[(size_t)h * 128 * 128 + (size_t)j * 128 + i] = (bf16_t)(s * norm1);
  }
}

// ---------------- Precompute: W_fk = norm2 * w_fc^T @ w_k2 (stored transposed 128x512) ---

__global__ __launch_bounds__(256) void precompute_fk(const float* wfc, const float* wk2, bf16_t* WfkT) {
  const int c0 = blockIdx.x * 32;
  __shared__ float swk2[64][128];
  __shared__ float swfc[64][32];
  float acc[16];
#pragma unroll
  for (int q = 0; q < 16; ++q) acc[q] = 0.f;
  const int obase = threadIdx.x * 16;
  for (int j0 = 0; j0 < 512; j0 += 64) {
    __syncthreads();
    for (int i = threadIdx.x; i < 64 * 128; i += 256)
      swk2[i >> 7][i & 127] = wk2[(size_t)(j0 + (i >> 7)) * 128 + (i & 127)];
    for (int i = threadIdx.x; i < 64 * 32; i += 256)
      swfc[i >> 5][i & 31] = wfc[(size_t)(j0 + (i >> 5)) * 512 + c0 + (i & 31)];
    __syncthreads();
    for (int jj = 0; jj < 64; ++jj) {
#pragma unroll
      for (int q = 0; q < 16; ++q) {
        int o = obase + q;
        acc[q] += swfc[jj][o >> 7] * swk2[jj][o & 127];
      }
    }
  }
  const float norm2 = 0.04419417382415922f; // 1/sqrt(512)
#pragma unroll
  for (int q = 0; q < 16; ++q) {
    int o = obase + q;
    int c = c0 + (o >> 7), d = o & 127;
    WfkT[(size_t)d * 512 + c] = (bf16_t)(acc[q] * norm2);
  }
}

// ---------------- Convert context fp32 -> bf16 --------------------------------

__global__ __launch_bounds__(256) void convert_ctx(const float* __restrict__ src,
                                                   bf16_t* __restrict__ dst, int n4) {
  int i = blockIdx.x * 256 + threadIdx.x;
  if (i < n4) {
    float4 v = ((const float4*)src)[i];
    dst[i * 4 + 0] = (bf16_t)v.x;
    dst[i * 4 + 1] = (bf16_t)v.y;
    dst[i * 4 + 2] = (bf16_t)v.z;
    dst[i * 4 + 3] = (bf16_t)v.w;
  }
}

// ---------------- Kernel 1: fused attention per (b, h) -------------------------
// T = st @ W_qk[h]; flash-softmax over n: S = T @ ctx^T (mask, online m/l),
// P += s @ ctx (via transposed LDS copy); U = (P/l) @ w_v1_h^T -> x_pre.

__global__ __launch_bounds__(128) void attn1_kernel(
    const float* __restrict__ state_t, const bf16_t* __restrict__ ctxd,
    const bf16_t* __restrict__ ctxr, const float* __restrict__ w_v1,
    const int* __restrict__ mask, const bf16_t* __restrict__ WqkT,
    bf16_t* __restrict__ xpre)
{
  const int b = blockIdx.x;
  const int h = blockIdx.y;
  const int tid  = threadIdx.x;
  const int w    = tid >> 5;
  const int lane = tid & 31;
  const int col  = lane & 15;
  const int rowg = lane >> 4;

  __shared__ bf16_t sT[64 * 128];            // 16 KB
  __shared__ bf16_t sC[2][CN * 128];         // 32 KB  row-major ctx chunks
  __shared__ bf16_t sCT[2][128 * CN];        // 32 KB  transposed ctx chunks
  __shared__ bf16_t sS[64 * CN];             // 8 KB   exp(scores)
  __shared__ unsigned char sMk[64 * CN];     // 4 KB

  const float* stb = state_t + (size_t)b * A_ * DIN;
  const bf16_t* gctx0 = ctxd + (size_t)b * N_ * DIN;
  const bf16_t* gctx1 = ctxr + (size_t)b * N_ * DIN;
  const int* mkb = mask + (size_t)b * A_ * N_;
  const bf16_t* wqk = WqkT + (size_t)h * 128 * 128;

  const int at = w;  // wave owns one 16-agent tile; at==0 is the depot tile

  // Phase 1: T = st @ W_qk (64x128); wave handles its agent tile, all 8 d-tiles
  {
    v8f accT[8];
#pragma unroll
    for (int dt = 0; dt < 8; ++dt) accT[dt] = zero8();
#pragma unroll
    for (int kf = 0; kf < 4; ++kf) {
      v16bf a = load_a_f32(stb + (size_t)(at * 16) * DIN + kf * 32, DIN, lane);
#pragma unroll
      for (int dt = 0; dt < 8; ++dt) {
        v16bf bb = load_b_row(wqk + (size_t)(dt * 16) * 128 + kf * 32, 128, lane);
        accT[dt] = wmma_bf(a, bb, accT[dt]);
      }
    }
#pragma unroll
    for (int dt = 0; dt < 8; ++dt)
#pragma unroll
      for (int v = 0; v < 8; ++v)
        sT[(at * 16 + rowg * 8 + v) * 128 + dt * 16 + col] = (bf16_t)accT[dt][v];
  }
  __syncthreads();

  // hoist T A-fragments (invariant over the whole n loop)
  v16bf tfr[4];
#pragma unroll
  for (int kf = 0; kf < 4; ++kf)
    tfr[kf] = load_a(sT + (size_t)(at * 16) * 128 + kf * 32, 128, lane);

  const bf16_t* myctx  = (at == 0) ? sC[0] : sC[1];
  const bf16_t* myctxT = (at == 0) ? sCT[0] : sCT[1];
  v8f accP[8];
#pragma unroll
  for (int dt = 0; dt < 8; ++dt) accP[dt] = zero8();
  float runm[8], runl[8];
#pragma unroll
  for (int v = 0; v < 8; ++v) { runm[v] = -1e30f; runl[v] = 0.f; }

  for (int n0 = 0; n0 < N_; n0 += CN) {
    const int nv = (N_ - n0 < CN) ? (N_ - n0) : CN;  // 64 or 16

    // ---- row-major chunk copy: async-to-LDS if available, else dword copy ----
#if USE_ASYNC_LDS
    {
      const int nb = nv * 256;  // bytes per ctx chunk
      for (int off = tid * 16; off < nb; off += 128 * 16) {
        __builtin_amdgcn_global_load_async_to_lds_b128(
            (gptr128_t)((char*)gctx0 + (size_t)n0 * 256 + off),
            (lptr128_t)((char*)sC[0] + off), 0, 0);
        __builtin_amdgcn_global_load_async_to_lds_b128(
            (gptr128_t)((char*)gctx1 + (size_t)n0 * 256 + off),
            (lptr128_t)((char*)sC[1] + off), 0, 0);
      }
    }
#else
    {
      const int nInts = nv * 64;
      const int* s0 = (const int*)(gctx0 + (size_t)n0 * DIN);
      const int* s1 = (const int*)(gctx1 + (size_t)n0 * DIN);
      int* d0 = (int*)sC[0];
      int* d1 = (int*)sC[1];
      for (int i = tid; i < nInts; i += 128) { d0[i] = s0[i]; d1[i] = s1[i]; }
    }
#endif

    // ---- transposed copy: 4x4 bf16 block transpose from global (L2-hot) ----
    {
      const bf16_t* gs[2] = { gctx0 + (size_t)n0 * DIN, gctx1 + (size_t)n0 * DIN };
      const int nblk = (nv >> 2) * 32;
      for (int cxi = 0; cxi < 2; ++cxi) {
        const bf16_t* g = gs[cxi];
        bf16_t* td = sCT[cxi];
        for (int blk = tid; blk < nblk; blk += 128) {
          int nb4 = (blk >> 5) * 4;
          int d0  = (blk & 31) * 4;
          uint2 r0 = *(const uint2*)(g + (size_t)(nb4 + 0) * DIN + d0);
          uint2 r1 = *(const uint2*)(g + (size_t)(nb4 + 1) * DIN + d0);
          uint2 r2 = *(const uint2*)(g + (size_t)(nb4 + 2) * DIN + d0);
          uint2 r3 = *(const uint2*)(g + (size_t)(nb4 + 3) * DIN + d0);
          uint2 c;
          c.x = (r0.x & 0xffffu) | (r1.x << 16);
          c.y = (r2.x & 0xffffu) | (r3.x << 16);
          *(uint2*)(td + (size_t)(d0 + 0) * CN + nb4) = c;
          c.x = (r0.x >> 16) | (r1.x & 0xffff0000u);
          c.y = (r2.x >> 16) | (r3.x & 0xffff0000u);
          *(uint2*)(td + (size_t)(d0 + 1) * CN + nb4) = c;
          c.x = (r0.y & 0xffffu) | (r1.y << 16);
          c.y = (r2.y & 0xffffu) | (r3.y << 16);
          *(uint2*)(td + (size_t)(d0 + 2) * CN + nb4) = c;
          c.x = (r0.y >> 16) | (r1.y & 0xffff0000u);
          c.y = (r2.y >> 16) | (r3.y & 0xffff0000u);
          *(uint2*)(td + (size_t)(d0 + 3) * CN + nb4) = c;
        }
      }
      if (nv < CN) { // zero n-pad cols 16..31 of sCT for the padded K=32 frag
        for (int i = tid; i < 128 * 4; i += 128) {
          int row = i >> 2, q = i & 3;
          uint2 z; z.x = 0u; z.y = 0u;
          *(uint2*)(sCT[0] + (size_t)row * CN + 16 + q * 4) = z;
          *(uint2*)(sCT[1] + (size_t)row * CN + 16 + q * 4) = z;
        }
      }
    }

    // ---- mask chunk -> bytes ----
    {
      const int sh = (nv == CN) ? 6 : 4;
      for (int i = tid; i < 64 * nv; i += 128) {
        int a = i >> sh, n = i & (nv - 1);
        sMk[a * CN + n] = (mkb[(size_t)a * N_ + n0 + n] != 0) ? 1 : 0;
      }
    }
#if USE_ASYNC_LDS
    __builtin_amdgcn_s_wait_asynccnt(0);
#endif
    __syncthreads();

    const int ntile = nv >> 4;          // 4 or 1
    const int kkmax = (nv + 31) >> 5;   // 2 or 1

    // S = T @ ctx^T for this agent tile
    float u[4][8];
    for (int nt = 0; nt < ntile; ++nt) {
      v8f s = zero8();
#pragma unroll
      for (int kf = 0; kf < 4; ++kf) {
        v16bf bb = load_b_row(myctx + (size_t)(nt * 16) * 128 + kf * 32, 128, lane);
        s = wmma_bf(tfr[kf], bb, s);
      }
#pragma unroll
      for (int v = 0; v < 8; ++v) {
        int m = rowg * 8 + v;
        u[nt][v] = sMk[(at * 16 + m) * CN + nt * 16 + col] ? -10000.0f : s[v];
      }
    }

    // online softmax update (row-consistent: reduce across the 16-lane group)
    float alpha[8];
#pragma unroll
    for (int v = 0; v < 8; ++v) {
      float tm = u[0][v];
      for (int nt = 1; nt < ntile; ++nt) tm = fmaxf(tm, u[nt][v]);
#pragma unroll
      for (int off = 1; off < 16; off <<= 1) tm = fmaxf(tm, __shfl_xor(tm, off, 32));
      float nm = fmaxf(runm[v], tm);
      alpha[v] = __expf(runm[v] - nm);
      runm[v]  = nm;
      float psum = 0.f;
      for (int nt = 0; nt < ntile; ++nt) {
        float p = __expf(u[nt][v] - nm);
        psum += p;
        sS[(at * 16 + rowg * 8 + v) * CN + nt * 16 + col] = (bf16_t)p;
      }
#pragma unroll
      for (int off = 1; off < 16; off <<= 1) psum += __shfl_xor(psum, off, 32);
      runl[v] = runl[v] * alpha[v] + psum;
    }
    if (nv < CN) { // zero s-pad cols 16..31 for the K=32 frag
#pragma unroll
      for (int v = 0; v < 8; ++v)
        sS[(at * 16 + rowg * 8 + v) * CN + 16 + col] = (bf16_t)0.0f;
    }
#pragma unroll
    for (int dt = 0; dt < 8; ++dt)
#pragma unroll
      for (int v = 0; v < 8; ++v) accP[dt][v] *= alpha[v];

    // P += s @ ctx_chunk  (B frags contiguous from transposed copy)
    for (int kk = 0; kk < kkmax; ++kk) {
      v16bf a = load_a(sS + (size_t)(at * 16) * CN + kk * 32, CN, lane);
#pragma unroll
      for (int dt = 0; dt < 8; ++dt) {
        v16bf bb = load_b_row(myctxT + (size_t)(dt * 16) * CN + kk * 32, CN, lane);
        accP[dt] = wmma_bf(a, bb, accP[dt]);
      }
    }
    __syncthreads();
  }

  // finalize: Pn = P / l (bf16 in LDS), then U = Pn @ w_v1_h^T -> x_pre
  bf16_t* sPn = sC[0];  // reuse
  float invl[8];
#pragma unroll
  for (int v = 0; v < 8; ++v) invl[v] = 1.0f / runl[v];
#pragma unroll
  for (int dt = 0; dt < 8; ++dt)
#pragma unroll
    for (int v = 0; v < 8; ++v)
      sPn[(at * 16 + rowg * 8 + v) * 128 + dt * 16 + col] = (bf16_t)(accP[dt][v] * invl[v]);

  for (int ut = 0; ut < 4; ++ut) {
    v8f accU = zero8();
#pragma unroll
    for (int kf = 0; kf < 4; ++kf) {
      v16bf a  = load_a(sPn + (size_t)(at * 16) * 128 + kf * 32, 128, lane);
      v16bf bb = load_b_row_f32(w_v1 + (size_t)(h * 64 + ut * 16) * DIN + kf * 32, DIN, lane);
      accU = wmma_bf(a, bb, accU);
    }
#pragma unroll
    for (int v = 0; v < 8; ++v) {
      int m = rowg * 8 + v;
      int d = ut * 16 + col;
      size_t off;
      if (at == 0) { // depot block: (H,16,64) contiguous
        off = (size_t)b * 32768 + (size_t)h * 1024 + (size_t)m * 64 + d;
      } else {       // rest block: (H,48,64) contiguous after depot's 8192
        int ar = at * 16 + m - 16;
        off = (size_t)b * 32768 + 8192 + (size_t)h * 3072 + (size_t)ar * 64 + d;
      }
      xpre[off] = (bf16_t)accU[v];
    }
  }
}

// ---------------- Kernel 2: y = x_pre @ W_fk  (64x512 @ 512x128 per batch) ------

__global__ __launch_bounds__(128) void fc_k2_kernel(const bf16_t* __restrict__ xpre,
                                                    const bf16_t* __restrict__ WfkT,
                                                    bf16_t* __restrict__ y)
{
  const int b = blockIdx.x;
  const int lane = threadIdx.x & 31;
  const int at   = threadIdx.x >> 5;
  const int col  = lane & 15;
  const int rowg = lane >> 4;
  const bf16_t* xb = xpre + (size_t)b * 32768;
  v8f acc[8];
#pragma unroll
  for (int dt = 0; dt < 8; ++dt) acc[dt] = zero8();
  for (int kf = 0; kf < 16; ++kf) {
    v16bf a = load_a(xb + (size_t)(at * 16) * 512 + kf * 32, 512, lane);
#pragma unroll
    for (int dt = 0; dt < 8; ++dt) {
      v16bf bb = load_b_row(WfkT + (size_t)(dt * 16) * 512 + kf * 32, 512, lane);
      acc[dt] = wmma_bf(a, bb, acc[dt]);
    }
  }
#pragma unroll
  for (int dt = 0; dt < 8; ++dt)
#pragma unroll
    for (int v = 0; v < 8; ++v)
      y[(size_t)b * 8192 + (size_t)(at * 16 + rowg * 8 + v) * 128 + dt * 16 + col] =
          (bf16_t)acc[dt][v];
}

// ---------------- Kernel 3: comp = y @ ctx^T, mask, softmax (recompute trick) ---

__global__ __launch_bounds__(128) void score_kernel(
    const bf16_t* __restrict__ y, const bf16_t* __restrict__ ctxd,
    const bf16_t* __restrict__ ctxr, const int* __restrict__ mask,
    const int* __restrict__ Tptr, float* __restrict__ out)
{
  const int b = blockIdx.x, at = blockIdx.y;
  const int w    = threadIdx.x >> 5;
  const int lane = threadIdx.x & 31;
  const int col  = lane & 15;
  const int rowg = lane >> 4;
  __shared__ float sM[4][16], sZ[4][16], sGM[16], sGZi[16];

  const bf16_t* ctx = ((at == 0) ? ctxd : ctxr) + (size_t)b * N_ * DIN;
  const bf16_t* yb  = y + (size_t)b * 8192 + (size_t)(at * 16) * 128;
  const int* mkb    = mask + ((size_t)b * A_ + at * 16) * N_;
  const float invT  = 1.0f / (float)(*Tptr);

  v16bf afr[4];
#pragma unroll
  for (int kf = 0; kf < 4; ++kf) afr[kf] = load_a(yb + kf * 32, 128, lane);

  // lane-local online (max, sum-exp); cross-lane merge deferred to the end
  float lm[8], lz[8];
#pragma unroll
  for (int v = 0; v < 8; ++v) { lm[v] = -1e30f; lz[v] = 0.f; }

  for (int nt = w; nt < 125; nt += 4) {
    v8f s = zero8();
#pragma unroll
    for (int kf = 0; kf < 4; ++kf) {
      v16bf bb = load_b_row(ctx + (size_t)(nt * 16) * DIN + kf * 32, DIN, lane);
      s = wmma_bf(afr[kf], bb, s);
    }
#pragma unroll
    for (int v = 0; v < 8; ++v) {
      int m = rowg * 8 + v;
      float lg = (mkb[(size_t)m * N_ + nt * 16 + col] != 0) ? -10000.0f : s[v];
      lg *= invT;
      float nm = fmaxf(lm[v], lg);
      lz[v] = lz[v] * __expf(lm[v] - nm) + __expf(lg - nm);
      lm[v] = nm;
    }
  }
  // cross-lane (m,z) pair merge within each 16-lane row group
#pragma unroll
  for (int v = 0; v < 8; ++v) {
#pragma unroll
    for (int off = 1; off < 16; off <<= 1) {
      float om = __shfl_xor(lm[v], off, 32);
      float oz = __shfl_xor(lz[v], off, 32);
      float nm = fmaxf(lm[v], om);
      lz[v] = lz[v] * __expf(lm[v] - nm) + oz * __expf(om - nm);
      lm[v] = nm;
    }
  }
  if (col == 0) {
#pragma unroll
    for (int v = 0; v < 8; ++v) {
      sM[w][rowg * 8 + v] = lm[v];
      sZ[w][rowg * 8 + v] = lz[v];
    }
  }
  __syncthreads();
  if (threadIdx.x < 16) {
    float gm = sM[0][threadIdx.x];
    for (int ww = 1; ww < 4; ++ww) gm = fmaxf(gm, sM[ww][threadIdx.x]);
    float gz = 0.f;
    for (int ww = 0; ww < 4; ++ww) gz += sZ[ww][threadIdx.x] * __expf(sM[ww][threadIdx.x] - gm);
    sGM[threadIdx.x]  = gm;
    sGZi[threadIdx.x] = 1.0f / gz;
  }
  __syncthreads();
  float gm[8], gzi[8];
#pragma unroll
  for (int v = 0; v < 8; ++v) { gm[v] = sGM[rowg * 8 + v]; gzi[v] = sGZi[rowg * 8 + v]; }

  // pass 2: recompute logits, write normalized softmax
  float* ob = out + ((size_t)b * A_ + at * 16) * N_;
  for (int nt = w; nt < 125; nt += 4) {
    v8f s = zero8();
#pragma unroll
    for (int kf = 0; kf < 4; ++kf) {
      v16bf bb = load_b_row(ctx + (size_t)(nt * 16) * DIN + kf * 32, DIN, lane);
      s = wmma_bf(afr[kf], bb, s);
    }
#pragma unroll
    for (int v = 0; v < 8; ++v) {
      int m = rowg * 8 + v;
      float lg = (mkb[(size_t)m * N_ + nt * 16 + col] != 0) ? -10000.0f : s[v];
      lg *= invT;
      ob[(size_t)m * N_ + nt * 16 + col] = __expf(lg - gm[v]) * gzi[v];
    }
  }
}

// ---------------- Host-side launch ---------------------------------------------

extern "C" void kernel_launch(void* const* d_in, const int* in_sizes, int n_in,
                              void* d_out, int out_size, void* d_ws, size_t ws_size,
                              hipStream_t stream) {
  (void)in_sizes; (void)n_in; (void)out_size; (void)ws_size;
  const float* state_t   = (const float*)d_in[0];
  const float* context   = (const float*)d_in[1];
  const float* context_d = (const float*)d_in[2];
  const float* w_q  = (const float*)d_in[3];
  const float* w_k1 = (const float*)d_in[4];
  const float* w_v1 = (const float*)d_in[5];
  const float* w_fc = (const float*)d_in[6];
  const float* w_k2 = (const float*)d_in[7];
  const int*   mask = (const int*)d_in[9];
  const int*   Tptr = (const int*)d_in[10];

  const size_t nctx = (size_t)B_ * N_ * DIN; // 32,768,000 elems per context
  bf16_t* ws     = (bf16_t*)d_ws;
  bf16_t* ctxd_b = ws;                                   // bf16 context_d
  bf16_t* ctxr_b = ctxd_b + nctx;                        // bf16 context
  bf16_t* xpre   = ctxr_b + nctx;                        // B*64*512
  bf16_t* yb     = xpre + (size_t)B_ * 64 * 512;         // B*64*128
  bf16_t* WqkT   = yb + (size_t)B_ * 64 * 128;           // 8*128*128
  bf16_t* WfkT   = WqkT + (size_t)8 * 128 * 128;         // 128*512

  precompute_qk<<<8, 256, 0, stream>>>(w_q, w_k1, WqkT);
  precompute_fk<<<16, 256, 0, stream>>>(w_fc, w_k2, WfkT);
  const int n4 = (int)(nctx / 4);
  convert_ctx<<<(n4 + 255) / 256, 256, 0, stream>>>(context_d, ctxd_b, n4);
  convert_ctx<<<(n4 + 255) / 256, 256, 0, stream>>>(context, ctxr_b, n4);
  attn1_kernel<<<dim3(B_, H_), 128, 0, stream>>>(state_t, ctxd_b, ctxr_b, w_v1, mask, WqkT, xpre);
  fc_k2_kernel<<<B_, 128, 0, stream>>>(xpre, WfkT, yb);
  score_kernel<<<dim3(B_, 4), 128, 0, stream>>>(yb, ctxd_b, ctxr_b, mask, Tptr, (float*)d_out);
}
